// AttentionBlock_85100482003173
// MI455X (gfx1250) — compile-verified, matrix operand order checked by reading, not run on hardware
//
#include <hip/hip_runtime.h>

typedef __attribute__((ext_vector_type(16))) _Float16 v16h;
typedef __attribute__((ext_vector_type(8)))  float    v8f;
typedef unsigned int u32x4 __attribute__((ext_vector_type(4)));
typedef int          i32x8 __attribute__((ext_vector_type(8)));
typedef int          i32x4 __attribute__((ext_vector_type(4)));

#define C_IN 256
#define HW   4096
#define CK   64
#define CV   256

#define LDX 264   // 256 + 8 half pad (kernel 1 xT staging)
#define LDK 72    // 64 + 8 half pad  (matches TDM pad: 4 DW per 32 DW)
#define LDV 72
#define LDP 72

#if __has_builtin(__builtin_amdgcn_tensor_load_to_lds) && __has_builtin(__builtin_amdgcn_s_wait_tensorcnt)
#define USE_TDM 1
#else
#define USE_TDM 0
#endif

union U16 { v16h v; uint4 u[2]; };

static __device__ __forceinline__ v8f wmma_f16(v16h a, v16h b, v8f c) {
  return __builtin_amdgcn_wmma_f32_16x16x32_f16(false, a, false, b, (short)0, c, false, false);
}

#if USE_TDM
// Issue one TDM 2-D tile load (f16 elements) global -> LDS, with LDS padding
// of 4 DWORDs (8 halfs) after every 32 DWORDs (64 halfs) stored, matching the
// 72-half padded row stride of the LDS tiles. Descriptor per CDNA5 ISA §8.
// This toolchain exposes the 6-argument builtin (groups 0..3 + 8-dword extra
// group + cache policy).
static __device__ __forceinline__ void tdm_load_2d_f16(
    unsigned lds_off, const void* gaddr,
    unsigned tile_d0, unsigned tile_d1,          // elements, rows
    unsigned d0_stride)                          // row stride in elements
{
  unsigned long long ga = (unsigned long long)(size_t)gaddr;
  u32x4 g0;
  g0[0] = 1u;                                    // count=1 (valid), no gather
  g0[1] = lds_off;                               // lds_addr (bytes)
  g0[2] = (unsigned)(ga & 0xffffffffu);          // global_addr[31:0]
  g0[3] = (unsigned)((ga >> 32) & 0x01ffffffu)   // global_addr[56:32]
        | (2u << 30);                            // type=2 ("image")
  i32x8 g1;
  g1[0] = (int)((1u << 16)                       // data_size = 1 -> 2 bytes
        |      (1u << 20)                        // pad_enable
        |      (4u << 22)                        // pad_interval: 32 DWORDs
        |      (3u << 25));                      // pad_amount: 4 DWORDs
  g1[1] = (int)((tile_d0 & 0xffffu) << 16);      // tensor_dim0 = tile_d0
  g1[2] = (int)((tile_d1 & 0xffffu) << 16);      // tensor_dim1 = tile_d1
  g1[3] = (int)((tile_d0 & 0xffffu) << 16);      // tile_dim0
  g1[4] = (int)(tile_d1 & 0xffffu);              // tile_dim1 (tile_dim2 = 0)
  g1[5] = (int)d0_stride;                        // tensor_dim0_stride[31:0]
  g1[6] = 0;                                     // stride[47:32] | d1 stride
  g1[7] = 0;
  i32x4 z4 = {0, 0, 0, 0};                       // groups 2/3: 2-D tensor
  i32x8 z8 = {0, 0, 0, 0, 0, 0, 0, 0};
  __builtin_amdgcn_tensor_load_to_lds(g0, g1, z4, z4, z8, 0);
}
#endif

// ---------------------------------------------------------------------------
// Kernel 1: fused 1x1-conv projections (K with BN folded, V), fp32 -> f16.
//   Kt[b][i][c]  (HW x 64, row-major)   : Q A-frags AND K B-frags contiguous
//   V [b][cv][j] (256 x HW, row-major)  : PV B-frags contiguous along j
// ---------------------------------------------------------------------------
__global__ __launch_bounds__(128) void proj_kernel(
    const float* __restrict__ x,  const float* __restrict__ Wk, const float* __restrict__ bk,
    const float* __restrict__ bn_gamma, const float* __restrict__ bn_beta,
    const float* __restrict__ bn_mean,  const float* __restrict__ bn_var,
    const float* __restrict__ Wv, const float* __restrict__ bv,
    _Float16* __restrict__ Kt, _Float16* __restrict__ V16)
{
  __shared__ alignas(16) _Float16 ldsX[64 * LDX];   // [p][c] f16, A-operand staging

  const int tid  = threadIdx.x;
  const int lane = tid & 31;
  const int wave = tid >> 5;
  const int b    = blockIdx.x >> 6;          // 64 p-tiles per batch
  const int p0   = (blockIdx.x & 63) << 6;   // 64 pixels per block
  const int mrow = lane & 15;
  const int hi   = lane >> 4;                // lane group select (wave32 halves)

  const float* xb = x + (size_t)b * C_IN * HW;

  // Stage X^T tile (64 p x 256 c) into LDS as f16 (transpose on the way in).
  for (int t = tid; t < 256 * 16; t += 128) {
    int c  = t >> 4;
    int pg = (t & 15) << 2;
    float4 xv = *(const float4*)(xb + (size_t)c * HW + p0 + pg);
    ldsX[(pg + 0) * LDX + c] = (_Float16)xv.x;
    ldsX[(pg + 1) * LDX + c] = (_Float16)xv.y;
    ldsX[(pg + 2) * LDX + c] = (_Float16)xv.z;
    ldsX[(pg + 3) * LDX + c] = (_Float16)xv.w;
  }
  __syncthreads();

  // A fragment (16-bit A 16x32 layout): lane<16 -> M=lane, K {0..7,16..23};
  // lane>=16 -> M=lane-16, K {8..15,24..31}. Two contiguous 16B LDS loads.
  auto loadA = [&](int m0, int kc) -> v16h {
    int m  = m0 + mrow;
    int co = kc * 32 + hi * 8;
    U16 a;
    a.u[0] = *(const uint4*)&ldsX[m * LDX + co];
    a.u[1] = *(const uint4*)&ldsX[m * LDX + co + 16];
    return a.v;
  };
  // B fragment (32x16, mirror of A with M<->N): per-lane run along K at fixed
  // column. W row-major [n][c] means per-lane reads are contiguous fp32.
  auto loadB = [&](const float* __restrict__ W, int n0, int kc) -> v16h {
    int n  = n0 + mrow;
    int co = kc * 32 + hi * 8;
    const float* wp = W + (size_t)n * C_IN + co;
    float4 w0 = *(const float4*)(wp +  0);
    float4 w1 = *(const float4*)(wp +  4);
    float4 w2 = *(const float4*)(wp + 16);
    float4 w3 = *(const float4*)(wp + 20);
    v16h r;
    r[0]=(_Float16)w0.x; r[1]=(_Float16)w0.y; r[2]=(_Float16)w0.z; r[3]=(_Float16)w0.w;
    r[4]=(_Float16)w1.x; r[5]=(_Float16)w1.y; r[6]=(_Float16)w1.z; r[7]=(_Float16)w1.w;
    r[8]=(_Float16)w2.x; r[9]=(_Float16)w2.y; r[10]=(_Float16)w2.z; r[11]=(_Float16)w2.w;
    r[12]=(_Float16)w3.x; r[13]=(_Float16)w3.y; r[14]=(_Float16)w3.z; r[15]=(_Float16)w3.w;
    return r;
  };

  // ---- K projection: this wave owns columns [wave*16, wave*16+16) of Ck=64.
  {
    const int n0 = wave * 16;
    v8f acc[4];
#pragma unroll
    for (int m = 0; m < 4; ++m)
#pragma unroll
      for (int e = 0; e < 8; ++e) acc[m][e] = 0.0f;

#pragma unroll
    for (int kc = 0; kc < 8; ++kc) {
      v16h bf = loadB(Wk, n0, kc);
#pragma unroll
      for (int m = 0; m < 4; ++m)
        acc[m] = wmma_f16(loadA(m * 16, kc), bf, acc[m]);
    }
    // BN folded epilogue: out = dot*inv + (bk*inv + beta - mean*inv)
    const int cc  = n0 + mrow;
    float inv = bn_gamma[cc] * rsqrtf(bn_var[cc] + 1e-5f);
    float add = bn_beta[cc] + (bk[cc] - bn_mean[cc]) * inv;
#pragma unroll
    for (int m = 0; m < 4; ++m)
#pragma unroll
      for (int r = 0; r < 8; ++r) {
        int row = p0 + m * 16 + r + hi * 8;
        Kt[((size_t)b * HW + row) * CK + cc] = (_Float16)(acc[m][r] * inv + add);
      }
  }

  // ---- V projection: this wave owns 4 column tiles of Cv=256.
#pragma unroll
  for (int nt = 0; nt < 4; ++nt) {
    const int n0 = (wave * 4 + nt) * 16;
    v8f acc[4];
#pragma unroll
    for (int m = 0; m < 4; ++m)
#pragma unroll
      for (int e = 0; e < 8; ++e) acc[m][e] = 0.0f;

#pragma unroll
    for (int kc = 0; kc < 8; ++kc) {
      v16h bf = loadB(Wv, n0, kc);
#pragma unroll
      for (int m = 0; m < 4; ++m)
        acc[m] = wmma_f16(loadA(m * 16, kc), bf, acc[m]);
    }
    const int cc = n0 + mrow;
    float bva = bv[cc];
#pragma unroll
    for (int m = 0; m < 4; ++m)
#pragma unroll
      for (int r = 0; r < 8; ++r) {
        int row = p0 + m * 16 + r + hi * 8;
        V16[((size_t)b * CV + cc) * HW + row] = (_Float16)(acc[m][r] + bva);
      }
  }
}

// ---------------------------------------------------------------------------
// Kernel 2: fused flash attention.  energy = K^T K, softmax, O = V * attn^T.
// Block = 128 threads (4 wave32), block owns 64 query rows (16 per wave).
// j-loop stages 64-wide K/V tiles into LDS via the Tensor Data Mover.
// ---------------------------------------------------------------------------
__global__ __launch_bounds__(128) void attn_kernel(
    const _Float16* __restrict__ Kt, const _Float16* __restrict__ V16,
    float* __restrict__ out)
{
  __shared__ alignas(16) _Float16 ldsK[64 * LDK];      // [j][c]
  __shared__ alignas(16) _Float16 ldsV[CV * LDV];      // [cv][j]
  __shared__ alignas(16) _Float16 ldsP[4 * 16 * LDP];  // per-wave [m][j]

  const int tid  = threadIdx.x;
  const int lane = tid & 31;
  const int wave = tid >> 5;
  const int b    = blockIdx.x >> 6;
  const int i0   = (blockIdx.x & 63) << 6;
  const int iw   = i0 + wave * 16;           // this wave's 16 query rows
  const int mrow = lane & 15;
  const int hi   = lane >> 4;

  const _Float16* Ktb = Kt  + (size_t)b * HW * CK;
  const _Float16* Vb  = V16 + (size_t)b * CV * HW;

#if USE_TDM
  const unsigned ldsK_off = (unsigned)(size_t)(const void*)&ldsK[0];
  const unsigned ldsV_off = (unsigned)(size_t)(const void*)&ldsV[0];
#endif

  // Q A-fragments (rows iw..iw+15, c = 0..63): two contiguous 16B loads each.
  U16 qa0, qa1;
  {
    const _Float16* qp = Ktb + (size_t)(iw + mrow) * CK;
    int co = hi * 8;
    qa0.u[0] = *(const uint4*)(qp + co);
    qa0.u[1] = *(const uint4*)(qp + co + 16);
    qa1.u[0] = *(const uint4*)(qp + 32 + co);
    qa1.u[1] = *(const uint4*)(qp + 32 + co + 16);
  }

  v8f oacc[16];                         // 16 rows x 256 cv accumulator (f32)
#pragma unroll
  for (int ct = 0; ct < 16; ++ct)
#pragma unroll
    for (int e = 0; e < 8; ++e) oacc[ct][e] = 0.0f;

  float mrw[8], lrw[8];
#pragma unroll
  for (int r = 0; r < 8; ++r) { mrw[r] = -3.0e38f; lrw[r] = 0.0f; }

  _Float16* myP = ldsP + wave * 16 * LDP;

  for (int j0 = 0; j0 < HW; j0 += 64) {
    __syncthreads();                    // previous tile fully consumed
#if USE_TDM
    // Tensor Data Mover: one DMA per tile, LDS padding done by descriptor.
    if (wave == 0) {
      tdm_load_2d_f16(ldsK_off, Ktb + (size_t)j0 * CK, 64, 64, CK);
      tdm_load_2d_f16(ldsV_off, Vb + j0, 64, CV, HW);
      __builtin_amdgcn_s_wait_tensorcnt(0);
    }
#else
    // Fallback: cooperative vector-load staging.
    for (int t = tid; t < 256; t += 128) {
      int j = t >> 2, ch = (t & 3) << 3;
      *(uint4*)&ldsK[j * LDK + ch] = *(const uint4*)(Ktb + (size_t)(j0 + j) * CK + ch);
    }
    for (int t = tid; t < 2048; t += 128) {
      int cv = t >> 3, ch = (t & 7) << 3;
      *(uint4*)&ldsV[cv * LDV + ch] = *(const uint4*)(Vb + (size_t)cv * HW + j0 + ch);
    }
#endif
    if (j0 + 64 < HW) {                 // hint next tile toward L2
      __builtin_prefetch(Ktb + (size_t)(j0 + 64) * CK + (size_t)tid * 64, 0, 1);
      __builtin_prefetch(Vb + (size_t)(tid) * HW + j0 + 64, 0, 1);
      __builtin_prefetch(Vb + (size_t)(tid + 128) * HW + j0 + 64, 0, 1);
    }
    __syncthreads();

    // S = Q * K_tile  (4 subtiles of 16 cols, K-dim = 64 -> 2 WMMAs each)
    v8f s[4];
#pragma unroll
    for (int jt = 0; jt < 4; ++jt) {
#pragma unroll
      for (int e = 0; e < 8; ++e) s[jt][e] = 0.0f;
      int jj = jt * 16 + mrow;
#pragma unroll
      for (int kc = 0; kc < 2; ++kc) {
        U16 bf;
        int co = kc * 32 + hi * 8;
        bf.u[0] = *(const uint4*)&ldsK[jj * LDK + co];
        bf.u[1] = *(const uint4*)&ldsK[jj * LDK + co + 16];
        s[jt] = wmma_f16(kc == 0 ? qa0.v : qa1.v, bf.v, s[jt]);
      }
    }

    // Online softmax: row reductions across 16-lane halves of the wave32.
#pragma unroll
    for (int r = 0; r < 8; ++r) {
      float t0 = fmaxf(fmaxf(s[0][r], s[1][r]), fmaxf(s[2][r], s[3][r]));
#pragma unroll
      for (int d = 1; d < 16; d <<= 1) t0 = fmaxf(t0, __shfl_xor(t0, d, 32));
      float mnew  = fmaxf(mrw[r], t0);
      float alpha = __expf(mrw[r] - mnew);
      mrw[r] = mnew;
      float rs = 0.0f;
#pragma unroll
      for (int jt = 0; jt < 4; ++jt) {
        float p = __expf(s[jt][r] - mnew);
        s[jt][r] = p;
        rs += p;
      }
#pragma unroll
      for (int d = 1; d < 16; d <<= 1) rs += __shfl_xor(rs, d, 32);
      lrw[r] = lrw[r] * alpha + rs;
#pragma unroll
      for (int ct = 0; ct < 16; ++ct) oacc[ct][r] *= alpha;
    }

    // Re-layout P (C-layout) -> A-fragments via wave-private LDS.
#pragma unroll
    for (int jt = 0; jt < 4; ++jt)
#pragma unroll
      for (int r = 0; r < 8; ++r)
        myP[(r + hi * 8) * LDP + jt * 16 + mrow] = (_Float16)s[jt][r];

    U16 ap0, ap1;
    {
      const _Float16* pp = myP + mrow * LDP;
      int co = hi * 8;
      ap0.u[0] = *(const uint4*)(pp + co);
      ap0.u[1] = *(const uint4*)(pp + co + 16);
      ap1.u[0] = *(const uint4*)(pp + 32 + co);
      ap1.u[1] = *(const uint4*)(pp + 32 + co + 16);
    }

    // O += P * V_tile^T  (16 cv-tiles x 2 k-steps)
#pragma unroll
    for (int ct = 0; ct < 16; ++ct) {
      int cvv = ct * 16 + mrow;
#pragma unroll
      for (int kc = 0; kc < 2; ++kc) {
        U16 bf;
        int co = kc * 32 + hi * 8;
        bf.u[0] = *(const uint4*)&ldsV[cvv * LDV + co];
        bf.u[1] = *(const uint4*)&ldsV[cvv * LDV + co + 16];
        oacc[ct] = wmma_f16(kc == 0 ? ap0.v : ap1.v, bf.v, oacc[ct]);
      }
    }
  }

  // Epilogue: out[b][cv][i] = O / l
  float* ob = out + (size_t)b * CV * HW;
#pragma unroll
  for (int r = 0; r < 8; ++r) {
    float invl = 1.0f / lrw[r];
    int row = iw + r + hi * 8;
#pragma unroll
    for (int ct = 0; ct < 16; ++ct) {
      int cv = ct * 16 + mrow;
      ob[(size_t)cv * HW + row] = oacc[ct][r] * invl;
    }
  }
}

// ---------------------------------------------------------------------------
extern "C" void kernel_launch(void* const* d_in, const int* in_sizes, int n_in,
                              void* d_out, int out_size, void* d_ws, size_t ws_size,
                              hipStream_t stream) {
  (void)in_sizes; (void)n_in; (void)out_size; (void)ws_size;
  const float* x        = (const float*)d_in[0];
  const float* Wk       = (const float*)d_in[1];
  const float* bk       = (const float*)d_in[2];
  const float* bn_gamma = (const float*)d_in[3];
  const float* bn_beta  = (const float*)d_in[4];
  const float* bn_mean  = (const float*)d_in[5];
  const float* bn_var   = (const float*)d_in[6];
  const float* Wv       = (const float*)d_in[7];
  const float* bv       = (const float*)d_in[8];
  float* out = (float*)d_out;

  _Float16* Kt  = (_Float16*)d_ws;                    // 4*4096*64  f16 = 2 MB
  _Float16* V16 = Kt + (size_t)4 * HW * CK;           // 4*256*4096 f16 = 8 MB

  proj_kernel<<<dim3(4 * (HW / 64)), dim3(128), 0, stream>>>(
      x, Wk, bk, bn_gamma, bn_beta, bn_mean, bn_var, Wv, bv, Kt, V16);
  attn_kernel<<<dim3(4 * (HW / 64)), dim3(128), 0, stream>>>(Kt, V16, out);
}
